// VectorQuantize_18605798326356
// MI455X (gfx1250) — matrix operand order repaired and checked by reading, not compiled, not run on hardware
//
#include <hip/hip_runtime.h>
#include <hip/hip_bf16.h>

// Problem constants (from reference setup_inputs)
#define B_ 8
#define D_ 256
#define T_ 2048
#define K_ 8192
#define N_ (B_ * T_)          // 16384 query rows
#define ROWS 64               // rows handled per block (4 waves x 16 rows)
#define NTHREADS 128          // 4 wave32
#define CT_TOTAL (K_ / 16)    // 512 code tiles of 16 codes (even -> unroll x2 OK)
#define XS_STRIDE 264         // padded LDS row stride in bf16 elems (528B, 16B aligned, bank-skewed)
#define EPS_ 1e-5f

typedef __attribute__((ext_vector_type(16))) __bf16 v16bf;
typedef __attribute__((ext_vector_type(8)))  float  v8f;

struct alignas(16) U4x { unsigned int v[4]; };
struct alignas(32) U8x { U4x lo, hi; };

__device__ __forceinline__ unsigned short f2bfbits(float f) {
  union { float f; unsigned int u; } a; a.f = f;
  unsigned int r = a.u + 0x7FFFu + ((a.u >> 16) & 1u);  // round-to-nearest-even
  return (unsigned short)(r >> 16);
}

__device__ __forceinline__ v16bf pack16(U4x lo, U4x hi) {
  U8x p; p.lo = lo; p.hi = hi;
  return __builtin_bit_cast(v16bf, p);
}

// LDS byte address of a shared-memory pointer: generic (flat) addresses in the
// LDS aperture carry the workgroup-relative LDS offset in their low 32 bits
// (ISA 10.2).
__device__ __forceinline__ unsigned lds_addr(const void* p) {
  return (unsigned)(unsigned long long)p;
}

// Async global->LDS copy of one 16-byte chunk (CDNA5 GLOBAL_LOAD_ASYNC_TO_LDS,
// GVS addressing: mem = SGPR base + VGPR offset). Tracked by ASYNCcnt.
__device__ __forceinline__ void async_copy_b128(unsigned ldsDst, unsigned gOff,
                                                unsigned long long gBase) {
  asm volatile("global_load_async_to_lds_b128 %0, %1, %2 offset:0"
               :: "v"(ldsDst), "v"(gOff), "s"(gBase) : "memory");
}

__device__ __forceinline__ void wait_async0() {
  asm volatile("s_wait_asynccnt 0x0" ::: "memory");
}

// ---------------------------------------------------------------------------
// Kernel 1: codebook prep — bf16 copy of embeddings + per-code squared norms.
// One wave per code row (8 codes per 256-thread block).
// ---------------------------------------------------------------------------
__global__ __launch_bounds__(256) void vq_prep(const float* __restrict__ emb,
                                               float* __restrict__ enorm,
                                               unsigned short* __restrict__ ebf) {
  const int lane = threadIdx.x & 31;
  const int w    = threadIdx.x >> 5;
  const int k    = blockIdx.x * 8 + w;
  const float* row = emb + k * D_;
  const int d0 = lane * 8;
  float s = 0.0f;
  unsigned short h[8];
#pragma unroll
  for (int j = 0; j < 8; ++j) {
    float v = row[d0 + j];
    s += v * v;
    h[j] = f2bfbits(v);
  }
  U4x out;
#pragma unroll
  for (int j = 0; j < 4; ++j)
    out.v[j] = (unsigned int)h[2 * j] | ((unsigned int)h[2 * j + 1] << 16);
  *(U4x*)(ebf + k * D_ + d0) = out;
#pragma unroll
  for (int off = 16; off >= 1; off >>= 1) s += __shfl_xor(s, off, 32);
  if (lane == 0) enorm[k] = s;
}

// ---------------------------------------------------------------------------
// Kernel 2: fused argmin-GEMM + quantize + STE outputs + EMA accumulators.
// Block = 128 threads (4 waves), 64 rows. Each wave owns 16 rows; all waves
// share a double-buffered 16-code LDS tile filled by async global->LDS DMA.
// ---------------------------------------------------------------------------
__global__ __launch_bounds__(NTHREADS) void vq_main(
    const float* __restrict__ x, const float* __restrict__ emb,
    const float* __restrict__ enorm, const unsigned short* __restrict__ ebf,
    float* __restrict__ cs, float* __restrict__ dw,
    float* __restrict__ out1, float* __restrict__ out2, float* __restrict__ out3) {
  __shared__ alignas(16) unsigned short xs[ROWS * XS_STRIDE];     // x tile, bf16
  __shared__ alignas(16) unsigned short es[2][16 * XS_STRIDE];    // code tiles, bf16
  __shared__ float xnorm[ROWS];
  __shared__ int   sIdx[ROWS];

  const int tid  = threadIdx.x;
  const int lane = tid & 31;
  const int w    = tid >> 5;       // wave id 0..3
  const int col  = lane & 15;      // N index within tile / row-in-wave
  const int hi   = lane >> 4;      // lane half
  const int n0   = blockIdx.x * ROWS;
  const int b    = n0 / T_;
  const int t0   = n0 % T_;        // T_ % ROWS == 0 -> one batch per block
  const int bOff = b * D_ * T_;

  // Per-thread async staging slots: 8KB tile = 512 x b128, 128 threads x 4.
  // Tile ct is contiguous in ebf (16*256 bf16), so global offset is q*16.
  // Separate arrays per buffer so all indexing is compile-time (no v_movrels).
  unsigned ldsSlotA[4], ldsSlotB[4], gOffSlot[4];
#pragma unroll
  for (int i = 0; i < 4; ++i) {
    const int q    = tid + i * NTHREADS;  // 0..511
    const int code = q >> 5;              // 32 b128 chunks per code row
    const int e4   = q & 31;
    gOffSlot[i] = (unsigned)(q * 16);
    ldsSlotA[i] = lds_addr(&es[0][code * XS_STRIDE + e4 * 8]);
    ldsSlotB[i] = lds_addr(&es[1][code * XS_STRIDE + e4 * 8]);
  }
  const unsigned long long ebfBase = (unsigned long long)ebf;

  // Kick off async staging of tile 0 before doing anything else.
#pragma unroll
  for (int i = 0; i < 4; ++i) async_copy_b128(ldsSlotA[i], gOffSlot[i], ebfBase);

  // ---- stage x tile -> LDS bf16, accumulate row norms ----
  if (tid < ROWS) xnorm[tid] = 0.0f;
  __syncthreads();
  {
    const int m = tid & 63;   // t offset within tile (coalesced across lanes)
    const int g = tid >> 6;   // 0..1 -> d parity
    float acc = 0.0f;
    for (int d = g; d < D_; d += 2) {
      float v = x[bOff + d * T_ + t0 + m];
      acc += v * v;
      xs[m * XS_STRIDE + d] = f2bfbits(v);
    }
    atomicAdd(&xnorm[m], acc);
  }
  __syncthreads();

  // ---- hoist the wave's 8 A fragments (16x32 bf16 each) from LDS ----
  // A layout: lane holds row M=col; element i <-> K = (i&7) + 8*hi + 16*(i>=8)
  v16bf afrag[8];
  {
    const unsigned short* rowp = &xs[(w * 16 + col) * XS_STRIDE];
#pragma unroll
    for (int s = 0; s < 8; ++s) {
      const int d0 = s * 32 + hi * 8;
      afrag[s] = pack16(*(const U4x*)(rowp + d0), *(const U4x*)(rowp + d0 + 16));
    }
  }
  float xn[8];
#pragma unroll
  for (int r = 0; r < 8; ++r) xn[r] = xnorm[w * 16 + r + 8 * hi];

  float minv[8]; int mini[8];
#pragma unroll
  for (int r = 0; r < 8; ++r) { minv[r] = 3.4e38f; mini[r] = 0; }

  const unsigned short* browA = &es[0][col * XS_STRIDE];
  const unsigned short* browB = &es[1][col * XS_STRIDE];

  // One tile step: stage tile (ct+1) into the other buffer, run 8 WMMAs on the
  // current buffer with register-double-buffered B fragments, update argmin,
  // then wait async + barrier (skipped after the last tile).
  auto tile_step = [&](int ct, const unsigned short* brow, const unsigned* ldsNext) {
    if (ct + 1 < CT_TOTAL) {
      const unsigned long long gbase =
          ebfBase + (unsigned long long)(ct + 1) * (16 * D_ * 2);
#pragma unroll
      for (int i = 0; i < 4; ++i) async_copy_b128(ldsNext[i], gOffSlot[i], gbase);
    }

    const int   k  = ct * 16 + col;
    const float en = enorm[k];

    // B layout: lane holds column N=col; element i <-> K = (i&7) + 8*hi + 16*(i>=8).
    U4x l0 = *(const U4x*)(brow + hi * 8);
    U4x l1 = *(const U4x*)(brow + hi * 8 + 16);
    v8f c = {0.f, 0.f, 0.f, 0.f, 0.f, 0.f, 0.f, 0.f};
#pragma unroll
    for (int s = 0; s < 8; ++s) {
      v16bf bfrag = pack16(l0, l1);
      if (s < 7) {
        const int d0 = (s + 1) * 32 + hi * 8;
        l0 = *(const U4x*)(brow + d0);
        l1 = *(const U4x*)(brow + d0 + 16);
      }
      c = __builtin_amdgcn_wmma_f32_16x16x32_bf16(false, afrag[s], false, bfrag,
                                                  (short)0, c, false, false);
    }

    // C layout: VGPR r, lane -> M = r + 8*hi, N = col
#pragma unroll
    for (int r = 0; r < 8; ++r) {
      float dist = xn[r] + en - 2.0f * c[r];
      if (dist < minv[r]) { minv[r] = dist; mini[r] = k; }
    }

    if (ct + 1 < CT_TOTAL) {
      wait_async0();     // next buffer fully landed (per wave)
      __syncthreads();   // ...for all waves; also fences this buffer's readers
    }
  };

  // Tile 0 must have landed in es[0] for every wave before first compute.
  wait_async0();
  __syncthreads();

  for (int ct = 0; ct < CT_TOTAL; ct += 2) {
    tile_step(ct,     browA, ldsSlotB);  // compute buf0, stage into buf1
    tile_step(ct + 1, browB, ldsSlotA);  // compute buf1, stage into buf0
  }

  // ---- per-row argmin reduce within each 16-lane half ----
#pragma unroll
  for (int r = 0; r < 8; ++r) {
    float v = minv[r]; int i = mini[r];
#pragma unroll
    for (int off = 8; off >= 1; off >>= 1) {
      float v2 = __shfl_xor(v, off, 32);
      int   i2 = __shfl_xor(i, off, 32);
      if (v2 < v || (v2 == v && i2 < i)) { v = v2; i = i2; }
    }
    if (col == 0) sIdx[w * 16 + r + 8 * hi] = i;
  }
  __syncthreads();

  // ---- outputs: quantized (x2), passthrough x, EMA accumulators ----
  {
    const int m   = tid & 63;
    const int g   = tid >> 6;
    const int t   = t0 + m;
    const int idx = sIdx[m];
    const float* erow  = emb + idx * D_;
    float*       dwrow = dw  + idx * D_;
    for (int d = g; d < D_; d += 2) {
      const int o = bOff + d * T_ + t;   // coalesced across lanes (consecutive t)
      const float q  = erow[d];
      const float xv = x[o];
      out1[o] = q;
      out2[o] = xv;
      out3[o] = q;
      atomicAdd(&dwrow[d], xv);
    }
    if (tid < ROWS) atomicAdd(&cs[sIdx[tid]], 1.0f);
  }
}

// ---------------------------------------------------------------------------
// Kernel 3: EMA finalize — new_emb[k,d] = dw[k,d] / ema_cs[k], in place.
// sum(cluster_size) == N exactly (every row votes once).
// ---------------------------------------------------------------------------
__global__ __launch_bounds__(256) void vq_finalize(float* __restrict__ newemb,
                                                   const float* __restrict__ cs) {
  const int k = blockIdx.x;
  const int d = threadIdx.x;
  const float c = cs[k];
  const float ema = (c + EPS_) / ((float)N_ + (float)K_ * EPS_) * (float)N_;
  const int o = k * D_ + d;
  newemb[o] = newemb[o] / ema;
}

// ---------------------------------------------------------------------------
extern "C" void kernel_launch(void* const* d_in, const int* in_sizes, int n_in,
                              void* d_out, int out_size, void* d_ws, size_t ws_size,
                              hipStream_t stream) {
  const float* x   = (const float*)d_in[0];   // [B, D, T]
  const float* emb = (const float*)d_in[1];   // [K, D]

  float* out  = (float*)d_out;
  float* out1 = out;                               // quantized [B,D,T]
  float* out2 = out1 + (size_t)B_ * D_ * T_;       // vq_in_bdt == x
  float* out3 = out2 + (size_t)B_ * D_ * T_;       // stop_grad(quantized)
  float* newe = out3 + (size_t)B_ * D_ * T_;       // new_embeddings (dw accumulator)

  char* ws = (char*)d_ws;
  float*          enorm = (float*)ws;                                          // K floats
  unsigned short* ebf   = (unsigned short*)(ws + (size_t)K_ * sizeof(float));  // K*D bf16
  float*          cs    = (float*)(ws + (size_t)K_ * sizeof(float)
                                      + (size_t)K_ * D_ * sizeof(unsigned short)); // K floats

  hipMemsetAsync(newe, 0, (size_t)K_ * D_ * sizeof(float), stream);
  hipMemsetAsync(cs, 0, (size_t)K_ * sizeof(float), stream);

  vq_prep<<<K_ / 8, 256, 0, stream>>>(emb, enorm, ebf);
  vq_main<<<N_ / ROWS, NTHREADS, 0, stream>>>(x, emb, enorm, ebf, cs, newe,
                                              out1, out2, out3);
  vq_finalize<<<K_, D_, 0, stream>>>(newe, cs);
}
